// FlashAttentionModule_3530463117614
// MI455X (gfx1250) — matrix operand order repaired and checked by reading, compiled, not run on hardware
//
#include <hip/hip_runtime.h>

#define B_  4
#define S_  4096
#define D_  1024
#define M_  (B_*S_)   // 16384 rows total (batch*seq)

typedef __bf16 bf16_t;
typedef __bf16 v16bf __attribute__((ext_vector_type(16)));
typedef __bf16 v8bf  __attribute__((ext_vector_type(8)));
typedef float  v8f   __attribute__((ext_vector_type(8)));
typedef unsigned int v4u __attribute__((ext_vector_type(4)));
typedef int v8i __attribute__((ext_vector_type(8)));
typedef int v4i __attribute__((ext_vector_type(4)));

__device__ __forceinline__ v16bf cat16(v8bf lo, v8bf hi) {
  return __builtin_shufflevector(lo, hi, 0,1,2,3,4,5,6,7,8,9,10,11,12,13,14,15);
}

// A-fragment (16x32 bf16, M x K). rowk0 = &A[row_of_lane][k0].
// Lanes 0-15 hold K 0..7 and 16..23; lanes 16-31 hold K 8..15 and 24..31.
__device__ __forceinline__ v16bf load_a_frag(const bf16_t* rowk0, int half) {
  const v8bf* p0 = (const v8bf*)(rowk0 + (half ? 8 : 0));
  const v8bf* p1 = (const v8bf*)(rowk0 + 16 + (half ? 8 : 0));
  return cat16(*p0, *p1);
}

// B-fragment (32x16 bf16, K x N), sourced K-contiguous per output column.
// colk0 = &Bt[col_of_lane][k0]. Lanes 0-15 hold K 0..7 and 16..23; lanes 16-31 hold K 8..15 and 24..31.
__device__ __forceinline__ v16bf load_b_frag(const bf16_t* colk0, int half) {
  const v8bf* p0 = (const v8bf*)(colk0 + (half ? 8 : 0));
  const v8bf* p1 = (const v8bf*)(colk0 + 16 + (half ? 8 : 0));
  return cat16(*p0, *p1);
}

__device__ __forceinline__ v8f wmma_bf16(v16bf a, v16bf b, v8f c) {
  return __builtin_amdgcn_wmma_f32_16x16x32_bf16(false, a, false, b, (short)0, c,
                                                 false, false);
}

// ---- Tensor Data Mover: 2D tile (tile_w x tile_h bf16 elems) global -> LDS ----
// D# per CDNA5 ISA §8.3/8.4. Tensor sized == tile (no OOB clipping); rows packed
// contiguously into LDS. Issue from ONE wave; completion via TENSORcnt.
// amdgpu-toolchain (clang-23) builtin arity: 6 args
//   (v4u group0, v8i group1, v4i group2, v4i group3, v8i aux, i32 cpol)
__device__ __forceinline__ void tdm_load_2d(unsigned lds_addr, const void* gptr,
                                            unsigned tile_w, unsigned tile_h,
                                            unsigned row_stride_elems) {
  unsigned long long ga = (unsigned long long)(uintptr_t)gptr;
  v4u g0;
  g0.x = 1u;                                                  // count=1, user D#
  g0.y = lds_addr;                                            // LDS byte address
  g0.z = (unsigned)ga;                                        // global_addr[31:0]
  g0.w = (unsigned)((ga >> 32) & 0x01FFFFFFu) | 0x80000000u;  // ga[56:32] | type=2
  v8i g1;
  g1[0] = (int)(1u << 16);              // data_size=1 (2 bytes/elem)
  g1[1] = (int)(tile_w << 16);          // tensor_dim0[15:0]  (bits 63:48)
  g1[2] = (int)(tile_h << 16);          // tensor_dim1[15:0]  (bits 95:80)
  g1[3] = (int)(tile_w << 16);          // tile_dim0          (bits 127:112)
  g1[4] = (int)tile_h;                  // tile_dim1          (bits 143:128)
  g1[5] = (int)row_stride_elems;        // tensor_dim0_stride[31:0] (bits 191:160)
  g1[6] = 0;
  g1[7] = 0;
  v4i gz4 = {0, 0, 0, 0};
  v8i gz8 = {0, 0, 0, 0, 0, 0, 0, 0};
  __builtin_amdgcn_tensor_load_to_lds(g0, g1, gz4, gz4, gz8, 0);
}

// ---------------- conversion kernels ----------------

__global__ void f32_to_bf16_kernel(const float* __restrict__ src,
                                   unsigned short* __restrict__ dst_, int n) {
  bf16_t* dst = (bf16_t*)dst_;
  int i = blockIdx.x * blockDim.x + threadIdx.x;
  int stride = gridDim.x * blockDim.x;
  for (; i < n; i += stride) dst[i] = (bf16_t)src[i];
}

// Wt[n][k] = W[k][n], bf16
__global__ void w_transpose_bf16_kernel(const float* __restrict__ w,
                                        unsigned short* __restrict__ wt_) {
  bf16_t* wt = (bf16_t*)wt_;
  int i = blockIdx.x * blockDim.x + threadIdx.x;
  int stride = gridDim.x * blockDim.x;
  for (; i < D_ * D_; i += stride) {
    int n = i / D_, k = i - n * D_;
    wt[i] = (bf16_t)w[k * D_ + n];
  }
}

// ---------------- QKV projection GEMM ----------------
// C[M_,D_] = A[M_,D_] * Bt[D_(out),D_(in)]^T, bf16 in, bf16 out.
// TDM double-buffered LDS tiles: wave0 DMAs the next 64x32 A/B tiles while all
// 4 waves run WMMAs on the current buffer.
// transposed!=0 stores C^T with layout [D_][M_] (used for V).
__global__ __launch_bounds__(128)
void gemm_bf16_kernel(const unsigned short* __restrict__ A_,
                      const unsigned short* __restrict__ Bt_,
                      unsigned short* __restrict__ C_, int transposed) {
  const bf16_t* A  = (const bf16_t*)A_;
  const bf16_t* Bt = (const bf16_t*)Bt_;
  bf16_t* C = (bf16_t*)C_;
  __shared__ __align__(16) bf16_t As[2][64 * 32];
  __shared__ __align__(16) bf16_t Bs[2][64 * 32];
  const int tid  = threadIdx.x;
  const int wave = tid >> 5, lane = tid & 31;
  const int half = lane >> 4, l16 = lane & 15;
  const size_t m0 = (size_t)blockIdx.y * 64;
  const size_t n0 = (size_t)blockIdx.x * 64;
  v8f acc[4] = {};

  if (wave == 0) {
    tdm_load_2d((unsigned)(uintptr_t)&As[0][0], &A[m0 * D_], 32, 64, D_);
    tdm_load_2d((unsigned)(uintptr_t)&Bs[0][0], &Bt[n0 * D_], 32, 64, D_);
    __builtin_amdgcn_s_wait_tensorcnt(0);
  }
  __syncthreads();

  for (int k0 = 0; k0 < D_; k0 += 32) {
    const int cur = (k0 >> 5) & 1;
    if (wave == 0 && (k0 + 32) < D_) {
      tdm_load_2d((unsigned)(uintptr_t)&As[cur ^ 1][0], &A[m0 * D_ + k0 + 32],
                  32, 64, D_);
      tdm_load_2d((unsigned)(uintptr_t)&Bs[cur ^ 1][0], &Bt[n0 * D_ + k0 + 32],
                  32, 64, D_);
    }
    v16bf af = load_a_frag(&As[cur][(wave * 16 + l16) * 32], half);
    #pragma unroll
    for (int t = 0; t < 4; t++) {
      v16bf bfrag = load_b_frag(&Bs[cur][(t * 16 + l16) * 32], half);
      acc[t] = wmma_bf16(af, bfrag, acc[t]);
    }
    if (wave == 0) __builtin_amdgcn_s_wait_tensorcnt(0);
    __syncthreads();
  }

  #pragma unroll
  for (int t = 0; t < 4; t++) {
    #pragma unroll
    for (int r = 0; r < 8; r++) {
      size_t m = m0 + wave * 16 + r + half * 8;
      size_t n = n0 + t * 16 + l16;
      bf16_t v = (bf16_t)acc[t][r];
      if (transposed) C[n * (size_t)M_ + m] = v;
      else            C[m * (size_t)D_ + n] = v;
    }
  }
}

// ---------------- flash attention ----------------
// Block: 4 waves, 16 query rows, one batch. Loop over 64-key tiles.
// Stage 1: wave w computes S[16q,16k] for keys [j0+16w, j0+16w+16) over full D.
// Stage 2: wave w accumulates O[16q, 256d] for d-slice [256w, 256w+256).
__global__ __launch_bounds__(128)
void flash_attn_kernel(const unsigned short* __restrict__ Q_,
                       const unsigned short* __restrict__ K_,
                       const unsigned short* __restrict__ Vt_,
                       float* __restrict__ out) {
  const bf16_t* Qb = (const bf16_t*)Q_;
  const bf16_t* Kb = (const bf16_t*)K_;
  const bf16_t* Vt = (const bf16_t*)Vt_;   // layout [D_][M_]
  __shared__ __align__(16) bf16_t Qs[16 * 1024];   // 32KB, loaded once via TDM
  __shared__ __align__(16) bf16_t Ps[16 * 64];     // probs tile, bf16
  __shared__ float wmaxs[4][16];
  __shared__ float wsums[4][16];
  __shared__ float mstate[16];
  __shared__ float lstate[16];

  const int tid  = threadIdx.x;
  const int wave = tid >> 5, lane = tid & 31;
  const int half = lane >> 4, l16 = lane & 15;
  const int b  = blockIdx.y;
  const int q0 = blockIdx.x * 16;
  const size_t row0 = (size_t)b * S_;

  // Q tile rows are contiguous in memory: one 16x1024 (=16384-elem) TDM copy.
  if (wave == 0) {
    tdm_load_2d((unsigned)(uintptr_t)&Qs[0], &Qb[(row0 + q0) * D_],
                16384, 1, 16384);
    __builtin_amdgcn_s_wait_tensorcnt(0);
  }
  if (tid < 16) { mstate[tid] = -1e30f; lstate[tid] = 0.0f; }
  v8f o[16] = {};
  __syncthreads();

  const float scale = 0.03125f;          // 1/sqrt(1024)
  for (int j0 = 0; j0 < S_; j0 += 64) {
    // ---- stage 1: scores for this wave's 16 keys over full D ----
    const bf16_t* krow = &Kb[(row0 + j0 + wave * 16 + l16) * D_];
    v8f s = {};
    #pragma unroll 4
    for (int d0 = 0; d0 < D_; d0 += 32) {
      v16bf af    = load_a_frag(&Qs[l16 * 1024 + d0], half);
      v16bf bfrag = load_b_frag(&krow[d0], half);
      s = wmma_bf16(af, bfrag, s);
    }
    if (j0 + 64 < S_)
      __builtin_prefetch(&Kb[(row0 + j0 + 64 + wave * 16 + l16) * D_], 0, 0);

    // per-wave row max (reduce across 16 lanes of each half-wave)
    float sv[8], rmax[8];
    #pragma unroll
    for (int r = 0; r < 8; r++) {
      sv[r] = s[r] * scale;
      float v = sv[r];
      #pragma unroll
      for (int mk = 1; mk < 16; mk <<= 1) v = fmaxf(v, __shfl_xor(v, mk, 32));
      rmax[r] = v;
    }
    if (l16 == 0) {
      #pragma unroll
      for (int r = 0; r < 8; r++) wmaxs[wave][r + half * 8] = rmax[r];
    }
    __syncthreads();   // B1: wave maxes visible

    // combine maxes, compute probs, rescale O, row sums
    float mnew[8], alpha[8], rsum[8];
    #pragma unroll
    for (int r = 0; r < 8; r++) {
      int row = r + half * 8;
      float mn = mstate[row];
      mn = fmaxf(mn, wmaxs[0][row]);
      mn = fmaxf(mn, wmaxs[1][row]);
      mn = fmaxf(mn, wmaxs[2][row]);
      mn = fmaxf(mn, wmaxs[3][row]);
      mnew[r]  = mn;
      alpha[r] = __expf(mstate[row] - mn);
      float p  = __expf(sv[r] - mn);
      Ps[row * 64 + wave * 16 + l16] = (bf16_t)p;
      float v = p;
      #pragma unroll
      for (int mk = 1; mk < 16; mk <<= 1) v += __shfl_xor(v, mk, 32);
      rsum[r] = v;
    }
    if (l16 == 0) {
      #pragma unroll
      for (int r = 0; r < 8; r++) wsums[wave][r + half * 8] = rsum[r];
    }
    #pragma unroll
    for (int t = 0; t < 16; t++)
      #pragma unroll
      for (int r = 0; r < 8; r++) o[t][r] *= alpha[r];
    __syncthreads();   // B2: Ps and wave sums visible

    // wave0 lanes 0/16 update running softmax state (read next iter after B1)
    if (wave == 0 && l16 == 0) {
      #pragma unroll
      for (int r = 0; r < 8; r++) {
        int row = r + half * 8;
        lstate[row] = lstate[row] * alpha[r] +
                      wsums[0][row] + wsums[1][row] + wsums[2][row] + wsums[3][row];
        mstate[row] = mnew[r];
      }
    }

    // ---- stage 2: O[16, d-slice] += P[16,64] @ V[64, d-slice] ----
    #pragma unroll
    for (int ks = 0; ks < 2; ks++) {
      v16bf pf = load_a_frag(&Ps[l16 * 64 + ks * 32], half);
      #pragma unroll
      for (int t = 0; t < 16; t++) {
        size_t d = (size_t)wave * 256 + t * 16 + l16;
        const bf16_t* vrow = &Vt[d * (size_t)M_ + row0 + j0 + ks * 32];
        v16bf bfrag = load_b_frag(vrow, half);
        o[t] = wmma_bf16(pf, bfrag, o[t]);
      }
    }
  }
  __syncthreads();

  #pragma unroll
  for (int r = 0; r < 8; r++) {
    int row = r + half * 8;
    float inv = 1.0f / lstate[row];
    #pragma unroll
    for (int t = 0; t < 16; t++) {
      int d = wave * 256 + t * 16 + l16;
      out[(row0 + q0 + row) * D_ + d] = o[t][r] * inv;
    }
  }
}

// ---------------- launch ----------------

extern "C" void kernel_launch(void* const* d_in, const int* in_sizes, int n_in,
                              void* d_out, int out_size, void* d_ws, size_t ws_size,
                              hipStream_t stream) {
  (void)in_sizes; (void)n_in; (void)out_size; (void)ws_size;
  const float* x  = (const float*)d_in[0];
  const float* Wq = (const float*)d_in[1];
  const float* Wk = (const float*)d_in[2];
  const float* Wv = (const float*)d_in[3];

  char* ws = (char*)d_ws;
  const size_t XB = (size_t)M_ * D_ * sizeof(unsigned short);  // 32 MB
  const size_t WB = (size_t)D_ * D_ * sizeof(unsigned short);  //  2 MB
  unsigned short* xb  = (unsigned short*)(ws);
  unsigned short* wqt = (unsigned short*)(ws + XB);
  unsigned short* wkt = (unsigned short*)(ws + XB + WB);
  unsigned short* wvt = (unsigned short*)(ws + XB + 2 * WB);
  unsigned short* Qb  = (unsigned short*)(ws + XB + 3 * WB);
  unsigned short* Kbf = (unsigned short*)(ws + 2 * XB + 3 * WB);
  unsigned short* Vtb = (unsigned short*)(ws + 3 * XB + 3 * WB);

  f32_to_bf16_kernel<<<4096, 256, 0, stream>>>(x, xb, M_ * D_);
  w_transpose_bf16_kernel<<<2048, 256, 0, stream>>>(Wq, wqt);
  w_transpose_bf16_kernel<<<2048, 256, 0, stream>>>(Wk, wkt);
  w_transpose_bf16_kernel<<<2048, 256, 0, stream>>>(Wv, wvt);

  dim3 ggrid(D_ / 64, M_ / 64);
  gemm_bf16_kernel<<<ggrid, 128, 0, stream>>>(xb, wqt, Qb, 0);
  gemm_bf16_kernel<<<ggrid, 128, 0, stream>>>(xb, wkt, Kbf, 0);
  gemm_bf16_kernel<<<ggrid, 128, 0, stream>>>(xb, wvt, Vtb, 1);  // V stored transposed

  dim3 agrid(S_ / 16, B_);
  flash_attn_kernel<<<agrid, 128, 0, stream>>>(Qb, Kbf, Vtb, (float*)d_out);
}
